// CnnCrfTagger_50895362457659
// MI455X (gfx1250) — compile-verified, matrix operand order checked by reading, not run on hardware
//
#include <hip/hip_runtime.h>
#include <hip/hip_bf16.h>
#include <stdint.h>

// ---------------------------------------------------------------------------
// CNN-CRF tagger forward (NLL scalar) for MI455X / gfx1250.
// Dense layers -> v_wmma_f32_16x16x32_f16 (f16 in, f32 accumulate).
// LDS staging via global_load_async_to_lds_b128 (ASYNCcnt path).
// B=256 S=512 V=128 T=16 E=64 F=128 C=256 H=256 K=3
// ---------------------------------------------------------------------------

typedef __attribute__((ext_vector_type(16))) _Float16 v16h;
typedef __attribute__((ext_vector_type(8)))  _Float16 v8h;
typedef __attribute__((ext_vector_type(8)))  float    v8f;

// ---------------------------------------------------------------------------
// Pack an f32 weight tensor W[taps][Cin][Cout] into WMMA B-fragment blobs.
// Blob (tap k, k-chunk kc, n-tile nt) holds 512 f16, lane-major:
//   blob[lane*16 + j] = W[k][ kc*32 + (lane>>4)*8 + (j&7) + ((j>>3)<<4) ]
//                        [ nt*16 + (lane&15) ]
// matching the ISA 16-bit 32x16 B-matrix VGPR layout.
// ---------------------------------------------------------------------------
__global__ void pack_wmma_b_kernel(const float* __restrict__ W,
                                   _Float16* __restrict__ Wp,
                                   int taps, int Cin, int Cout) {
  int total = taps * (Cin / 32) * (Cout / 16) * 512;
  int tid = blockIdx.x * blockDim.x + threadIdx.x;
  if (tid >= total) return;
  int j    = tid & 15;
  int lane = (tid >> 4) & 31;
  int blob = tid >> 9;
  int NT = Cout / 16, KC = Cin / 32;
  int nt = blob % NT;
  int kc = (blob / NT) % KC;
  int k  = blob / (NT * KC);
  int K   = kc * 32 + ((lane >> 4) << 3) + (j & 7) + ((j >> 3) << 4);
  int col = nt * 16 + (lane & 15);
  Wp[tid] = (_Float16)W[((long long)(k * Cin + K)) * Cout + col];
}

// ---------------------------------------------------------------------------
// Embedding gather + f32->f16 convert: X0[t, 0:64] = emb[seq[t], 0:64]
// ---------------------------------------------------------------------------
__global__ void embed_kernel(const int* __restrict__ seq,
                             const float* __restrict__ emb,
                             _Float16* __restrict__ X0, long long Ntok) {
  long long tid = blockIdx.x * (long long)blockDim.x + threadIdx.x;
  if (tid >= Ntok * 8) return;
  long long t = tid >> 3;
  int c = (int)(tid & 7);
  const float* src = emb + (long long)seq[t] * 64 + c * 8;
  v8h o;
#pragma unroll
  for (int r = 0; r < 8; ++r) o[r] = (_Float16)src[r];
  *(v8h*)(X0 + t * 64 + c * 8) = o;
}

// ---------------------------------------------------------------------------
// Fused (conv-as-GEMM | GEMM) + bias + optional ReLU, WMMA 16x16x32 f16.
// Block = 128 threads = 4 waves. Block computes 16 tokens x (4 n-tiles).
// LDS stages an 18-row haloed token tile (rows = token-1 .. token+16):
//   rows 1..16  : always in-range -> async global->LDS copy (ASYNCcnt)
//   rows 0 / 17 : sequence-edge halo, zero-masked synchronous path
// Requires S % 16 == 0.
// ---------------------------------------------------------------------------
template <int CIN, int COUT, int TAPS, bool RELU, bool OUTF32>
__global__ void gemm_wmma_kernel(const _Float16* __restrict__ X,
                                 const _Float16* __restrict__ Wp,
                                 const float* __restrict__ bias,
                                 void* __restrict__ Yv, int S) {
  constexpr int NT = COUT / 16;
  __shared__ _Float16 tile[18 * CIN];

  const int tid  = threadIdx.x;
  const int lane = tid & 31;
  const int wave = tid >> 5;
  const long long t0 = (long long)blockIdx.x * 16;
  const int s0 = (int)(t0 % S);

  constexpr int H8 = CIN / 8;  // 16B chunks per row

  // ---- interior rows 1..16: direct async global -> LDS (no VGPR bounce) ----
  for (int i = tid; i < 16 * H8; i += 128) {
    int row = 1 + i / H8, c8 = i % H8;
    uint32_t lds = (uint32_t)(uintptr_t)&tile[row * CIN + c8 * 8];
    const _Float16* gp = X + (t0 + row - 1) * (long long)CIN + c8 * 8;
    asm volatile("global_load_async_to_lds_b128 %0, %1, off"
                 :: "v"(lds), "v"(gp) : "memory");
  }
  // ---- halo rows 0 and 17 (conv layers only): zero outside the sequence ----
  if (TAPS == 3) {
    for (int i = tid; i < 2 * H8; i += 128) {
      int row = (i >= H8) ? 17 : 0, c8 = i % H8;
      int s = s0 + row - 1;
      v8h v = {};
      if (s >= 0 && s < S)
        v = *((const v8h*)(X + (t0 + row - 1) * (long long)CIN) + c8);
      *(v8h*)&tile[row * CIN + c8 * 8] = v;
    }
  }
  asm volatile("s_wait_asynccnt 0x0" ::: "memory");
  __syncthreads();

  const int nt = blockIdx.y * 4 + wave;
  if (nt >= NT) return;

  const int ksub = (lane >> 4) << 3;   // 0 or 8
  const int mrow = lane & 15;

  v8f acc = {};
#pragma unroll
  for (int k = 0; k < TAPS; ++k) {
    const int rowA = mrow + ((TAPS == 3) ? k : 1);
#pragma unroll
    for (int kc = 0; kc < CIN / 32; ++kc) {
      // A fragment: two 16B LDS loads, ISA 16-bit A layout
      const _Float16* ap = &tile[rowA * CIN + kc * 32 + ksub];
      v8h a0 = *(const v8h*)ap;
      v8h a1 = *(const v8h*)(ap + 16);
      v16h a = __builtin_shufflevector(a0, a1, 0, 1, 2, 3, 4, 5, 6, 7,
                                       8, 9, 10, 11, 12, 13, 14, 15);
      // B fragment: pre-packed, one 32B global load per lane
      const _Float16* bp =
          Wp + (((long long)(k * (CIN / 32) + kc) * NT + nt) << 9) + lane * 16;
      __builtin_prefetch(bp + 512, 0, 0);  // global_prefetch_b8 (next frag)
      v16h b = *(const v16h*)bp;
      acc = __builtin_amdgcn_wmma_f32_16x16x32_f16(
          false, a, false, b, (short)0, acc, false, false);
    }
  }

  // ---- epilogue: bias (+ReLU) and store ----
  const int N = lane & 15;
  const float bn = bias[nt * 16 + N];
#pragma unroll
  for (int r = 0; r < 8; ++r) {
    int M = r + ((lane >> 4) << 3);
    float v = acc[r] + bn;
    if (RELU) v = fmaxf(v, 0.f);
    long long oi = (t0 + M) * (long long)COUT + nt * 16 + N;
    if (OUTF32)
      ((float*)Yv)[oi] = v;
    else
      ((_Float16*)Yv)[oi] = (_Float16)v;
  }
}

// ---------------------------------------------------------------------------
// LayerNorm over C=256 (eps 1e-5), one wave per token, wave32 shuffles.
// ---------------------------------------------------------------------------
__global__ void layernorm_kernel(const _Float16* __restrict__ X,
                                 const float* __restrict__ g,
                                 const float* __restrict__ b,
                                 _Float16* __restrict__ Y, long long Ntok) {
  long long t = blockIdx.x * (long long)(blockDim.x >> 5) + (threadIdx.x >> 5);
  int lane = threadIdx.x & 31;
  if (t >= Ntok) return;
  v8h xv = *(const v8h*)(X + t * 256 + lane * 8);
  float f[8], s = 0.f, s2 = 0.f;
#pragma unroll
  for (int r = 0; r < 8; ++r) {
    f[r] = (float)xv[r];
    s += f[r];
    s2 += f[r] * f[r];
  }
#pragma unroll
  for (int off = 16; off; off >>= 1) {
    s  += __shfl_xor(s, off, 32);
    s2 += __shfl_xor(s2, off, 32);
  }
  float mu  = s * (1.f / 256.f);
  float var = s2 * (1.f / 256.f) - mu * mu;
  float inv = rsqrtf(var + 1e-5f);
  v8h yv;
#pragma unroll
  for (int r = 0; r < 8; ++r)
    yv[r] = (_Float16)((f[r] - mu) * inv * g[lane * 8 + r] + b[lane * 8 + r]);
  *(v8h*)(Y + t * 256 + lane * 8) = yv;
}

// ---------------------------------------------------------------------------
// em[t, 0] += 10000 where seq[t] == 0 (pad positions biased to pad tag)
// ---------------------------------------------------------------------------
__global__ void padbias_kernel(const int* __restrict__ seq,
                               float* __restrict__ em, long long Ntok) {
  long long t = blockIdx.x * (long long)blockDim.x + threadIdx.x;
  if (t < Ntok && seq[t] == 0) em[t * 16] += 10000.f;
}

// ---------------------------------------------------------------------------
// CRF NLL per sequence: one wave per batch row. Lane j (mod 16) owns tag j.
// Forward algorithm logZ + gold path score; nll[b] = logZ - num.
// ---------------------------------------------------------------------------
__global__ void crf_kernel(const float* __restrict__ em,
                           const int* __restrict__ labels,
                           const float* __restrict__ start,
                           const float* __restrict__ endv,
                           const float* __restrict__ trans,
                           float* __restrict__ nll, int Bn, int Sn) {
  int wave = threadIdx.x >> 5;
  int lane = threadIdx.x & 31;
  int b = blockIdx.x * (blockDim.x >> 5) + wave;
  if (b >= Bn) return;
  const int j = lane & 15;

  float tr[16];  // column j of transition matrix
#pragma unroll
  for (int i = 0; i < 16; ++i) tr[i] = trans[i * 16 + j];

  const float* eb = em + (long long)b * Sn * 16;
  float alpha = start[j] + eb[j];

  for (int s = 1; s < Sn; ++s) {
    float ej = eb[s * 16 + j];
    float av[16];
#pragma unroll
    for (int i = 0; i < 16; ++i) av[i] = __shfl(alpha, i, 32);
    float mx = -3.402823466e38f;
#pragma unroll
    for (int i = 0; i < 16; ++i) mx = fmaxf(mx, av[i] + tr[i]);
    float ss = 0.f;
#pragma unroll
    for (int i = 0; i < 16; ++i) ss += __expf(av[i] + tr[i] - mx);
    alpha = ej + mx + __logf(ss);
  }
  alpha += endv[j];

  // logsumexp over 16 tags (lanes 16..31 mirror lanes 0..15)
  float mx = alpha;
#pragma unroll
  for (int off = 8; off; off >>= 1) mx = fmaxf(mx, __shfl_xor(mx, off, 32));
  float se = __expf(alpha - mx);
#pragma unroll
  for (int off = 8; off; off >>= 1) se += __shfl_xor(se, off, 32);
  float logZ = mx + __logf(se);

  // gold path score (lanes stride over positions)
  const int* lb = labels + (long long)b * Sn;
  float acc = 0.f;
  for (int s = lane; s < Sn; s += 32) {
    int y = lb[s];
    acc += eb[s * 16 + y];
    if (s > 0) acc += trans[lb[s - 1] * 16 + y];
  }
#pragma unroll
  for (int off = 16; off; off >>= 1) acc += __shfl_xor(acc, off, 32);

  if (lane == 0) {
    float num = acc + start[lb[0]] + endv[lb[Sn - 1]];
    nll[b] = logZ - num;
  }
}

// ---------------------------------------------------------------------------
// Mean over B values -> scalar
// ---------------------------------------------------------------------------
__global__ void mean_kernel(const float* __restrict__ nll,
                            float* __restrict__ out, int n) {
  __shared__ float sd[256];
  int tid = threadIdx.x;
  sd[tid] = (tid < n) ? nll[tid] : 0.f;
  __syncthreads();
  for (int s = 128; s > 0; s >>= 1) {
    if (tid < s) sd[tid] += sd[tid + s];
    __syncthreads();
  }
  if (tid == 0) out[0] = sd[0] / (float)n;
}

// ---------------------------------------------------------------------------
// Host launcher
// ---------------------------------------------------------------------------
extern "C" void kernel_launch(void* const* d_in, const int* in_sizes, int n_in,
                              void* d_out, int out_size, void* d_ws,
                              size_t ws_size, hipStream_t stream) {
  (void)in_sizes; (void)n_in; (void)out_size; (void)ws_size;
  constexpr int Bn = 256, Sn = 512, En = 64, Fn = 128, Cn = 256, Hn = 256;
  const long long Ntok = (long long)Bn * Sn;

  const int*   seq    = (const int*)d_in[0];
  const int*   labels = (const int*)d_in[1];
  /* d_in[2] = true_lengths (unused: reference uses full mask) */
  const float* emb    = (const float*)d_in[3];
  const float* cw0    = (const float*)d_in[4];
  const float* cb0    = (const float*)d_in[5];
  const float* cw1    = (const float*)d_in[6];
  const float* cb1    = (const float*)d_in[7];
  const float* cw2    = (const float*)d_in[8];
  const float* cb2    = (const float*)d_in[9];
  const float* pw     = (const float*)d_in[10];
  const float* pb     = (const float*)d_in[11];
  const float* lng    = (const float*)d_in[12];
  const float* lnb    = (const float*)d_in[13];
  const float* f1w    = (const float*)d_in[14];
  const float* f1b    = (const float*)d_in[15];
  const float* f2w    = (const float*)d_in[16];
  const float* f2b    = (const float*)d_in[17];
  const float* cstart = (const float*)d_in[18];
  const float* cend   = (const float*)d_in[19];
  const float* ctrans = (const float*)d_in[20];

  // ---- workspace carve (~143 MB) ----
  char* ws = (char*)d_ws;
  size_t off = 0;
  auto carve = [&](size_t bytes) {
    void* p = ws + off;
    off += (bytes + 255) & ~(size_t)255;
    return p;
  };
  _Float16* bufA = (_Float16*)carve(Ntok * 256 * 2);
  _Float16* bufB = (_Float16*)carve(Ntok * 256 * 2);
  float*    em   = (float*)carve(Ntok * 16 * 4);
  _Float16* wp0  = (_Float16*)carve((size_t)3 * 2 * 8 * 512 * 2);   // conv0
  _Float16* wp1  = (_Float16*)carve((size_t)3 * 4 * 8 * 512 * 2);   // conv1
  _Float16* wp2  = (_Float16*)carve((size_t)3 * 4 * 8 * 512 * 2);   // conv2
  _Float16* wpp  = (_Float16*)carve((size_t)1 * 4 * 16 * 512 * 2);  // proj
  _Float16* wp3  = (_Float16*)carve((size_t)1 * 8 * 16 * 512 * 2);  // fc1
  _Float16* wp4  = (_Float16*)carve((size_t)1 * 8 * 1 * 512 * 2);   // fc2
  float*    nll  = (float*)carve(Bn * 4);

  auto packGrid = [](int taps, int cin, int cout) {
    return (taps * (cin / 32) * (cout / 16) * 512 + 255) / 256;
  };
  pack_wmma_b_kernel<<<packGrid(3, En, Fn), 256, 0, stream>>>(cw0, wp0, 3, En, Fn);
  pack_wmma_b_kernel<<<packGrid(3, Fn, Fn), 256, 0, stream>>>(cw1, wp1, 3, Fn, Fn);
  pack_wmma_b_kernel<<<packGrid(3, Fn, Fn), 256, 0, stream>>>(cw2, wp2, 3, Fn, Fn);
  pack_wmma_b_kernel<<<packGrid(1, Fn, Cn), 256, 0, stream>>>(pw,  wpp, 1, Fn, Cn);
  pack_wmma_b_kernel<<<packGrid(1, Cn, Hn), 256, 0, stream>>>(f1w, wp3, 1, Cn, Hn);
  pack_wmma_b_kernel<<<packGrid(1, Hn, 16), 256, 0, stream>>>(f2w, wp4, 1, Hn, 16);

  embed_kernel<<<(unsigned)((Ntok * 8 + 255) / 256), 256, 0, stream>>>(
      seq, emb, bufA, Ntok);

  const unsigned MT = (unsigned)(Ntok / 16);  // token tiles
  gemm_wmma_kernel<64, 128, 3, true, false>
      <<<dim3(MT, 2), 128, 0, stream>>>(bufA, wp0, cb0, bufB, Sn);
  gemm_wmma_kernel<128, 128, 3, true, false>
      <<<dim3(MT, 2), 128, 0, stream>>>(bufB, wp1, cb1, bufA, Sn);
  gemm_wmma_kernel<128, 128, 3, true, false>
      <<<dim3(MT, 2), 128, 0, stream>>>(bufA, wp2, cb2, bufB, Sn);
  gemm_wmma_kernel<128, 256, 1, false, false>
      <<<dim3(MT, 4), 128, 0, stream>>>(bufB, wpp, pb, bufA, Sn);

  layernorm_kernel<<<(unsigned)(Ntok / 8), 256, 0, stream>>>(
      bufA, lng, lnb, bufB, Ntok);

  gemm_wmma_kernel<256, 256, 1, true, false>
      <<<dim3(MT, 4), 128, 0, stream>>>(bufB, wp3, f1b, bufA, Sn);
  gemm_wmma_kernel<256, 16, 1, true, true>
      <<<dim3(MT, 1), 128, 0, stream>>>(bufA, wp4, f2b, em, Sn);

  padbias_kernel<<<(unsigned)((Ntok + 255) / 256), 256, 0, stream>>>(
      seq, em, Ntok);

  crf_kernel<<<Bn / 8, 256, 0, stream>>>(em, labels, cstart, cend, ctrans,
                                         nll, Bn, Sn);
  mean_kernel<<<1, 256, 0, stream>>>(nll, (float*)d_out, Bn);
}
